// EdgeMessageLayer_57449482551999
// MI455X (gfx1250) — matrix operand order repaired and checked by reading, not compiled, not run on hardware
//
#include <hip/hip_runtime.h>
#include <math.h>

#define Dd   128
#define NFc  16384
#define NEc  49152
#define Bc   4

typedef __attribute__((ext_vector_type(16))) __bf16        bf16x16;
typedef __attribute__((ext_vector_type(8)))  float         f32x8;
typedef __attribute__((ext_vector_type(8)))  unsigned int  u32x8;

__device__ __forceinline__ unsigned short f2bf(float x) {
  unsigned int u = __float_as_uint(x);
  u += 0x7FFFu + ((u >> 16) & 1u);           // round-to-nearest-even
  return (unsigned short)(u >> 16);
}
__device__ __forceinline__ unsigned int pack2(float a, float b) {
  return (unsigned int)f2bf(a) | ((unsigned int)f2bf(b) << 16);
}
// 16-bit A 16x32 (matches B): lanes 0-15 hold K0..15, lanes 16-31 hold K16..31 (2 per VGPR)
__device__ __forceinline__ int a_kloc(int j, int hi) {
  return hi * 16 + 2 * j;
}
// B 32x16: lane = column; lanes 0-15 hold K0..15, lanes 16-31 hold K16..31 (2 per VGPR)
__device__ __forceinline__ u32x8 load_bfrag(const unsigned short* p) {
  const uint4* q = (const uint4*)p;
  uint4 x = q[0], y = q[1];
  u32x8 r;
  r[0]=x.x; r[1]=x.y; r[2]=x.z; r[3]=x.w;
  r[4]=y.x; r[5]=y.y; r[6]=y.z; r[7]=y.w;
  return r;
}
__device__ __forceinline__ f32x8 wmma_bf16(u32x8 a, u32x8 b, f32x8 c) {
  return __builtin_amdgcn_wmma_f32_16x16x32_bf16(
      false, __builtin_bit_cast(bf16x16, a),
      false, __builtin_bit_cast(bf16x16, b),
      (short)0, c, false, false);
}
__device__ __forceinline__ float gelu_exact(float x) {
  return 0.5f * x * (1.0f + erff(x * 0.70710678118654752f));
}
__device__ __forceinline__ float sigmoidf_(float x) {
  return 1.0f / (1.0f + expf(-x));
}

// ---------------- weight transpose + bf16 convert: W[K][N] -> Wt[N][K] ----------------
__global__ void eml_wconv(const float* __restrict__ W, unsigned short* __restrict__ Wt,
                          int K, int N) {
  int i = blockIdx.x * blockDim.x + threadIdx.x;
  if (i >= K * N) return;
  int k = i / N, n = i % N;
  Wt[n * K + k] = f2bf(W[i]);
}

// ---------------- edge kernel: one wave (32 threads) per 16-edge tile ----------------
#define SMEM_BYTES 24832
__global__ __launch_bounds__(32) void eml_edge_kernel(
    const float* __restrict__ F, const float* __restrict__ E,
    const int* __restrict__ etf, const float* __restrict__ edge_mask,
    const unsigned short* __restrict__ Wt_fe1, const float* __restrict__ b_fe1,
    const unsigned short* __restrict__ Wt_fe2, const float* __restrict__ b_fe2,
    const unsigned short* __restrict__ Wt_ge,  const float* __restrict__ b_ge,
    const float* __restrict__ ln_g, const float* __restrict__ ln_b,
    float* __restrict__ E_out, float* __restrict__ face_msg, float* __restrict__ face_cnt)
{
  __shared__ __align__(16) unsigned char smem[SMEM_BYTES];
  unsigned short* sh_h = (unsigned short*)smem;      // 16x256 bf16 (8KB), reused as sh_x
  float* sh_x = (float*)smem;                        // 16x128 f32 alias
  float* sh_E = (float*)(smem + 8192);               // 16x128 f32
  float* sh_m = (float*)(smem + 16384);              // 16x128 f32
  int*   sh_f1 = (int*)(smem + 24576);
  int*   sh_f2 = (int*)(smem + 24640);
  float* sh_v  = (float*)(smem + 24704);

  const int tile = blockIdx.x;
  const int lane = threadIdx.x;
  const int lo = lane & 15, hi = lane >> 4;
  const int row0 = tile * 16;

  if (lane < 16) {
    int grow = row0 + lane;                 // flattened (B*NE)
    int b = grow / NEc, e = grow % NEc;
    int a0 = etf[(size_t)(b * NEc + e) * 2 + 0];
    int a1 = etf[(size_t)(b * NEc + e) * 2 + 1];
    float vm = (edge_mask[b * NEc + e] > 0.f && a0 >= 0 && a1 >= 0) ? 1.f : 0.f;
    int f1 = a0 < 0 ? 0 : (a0 > NFc - 1 ? NFc - 1 : a0);
    int f2 = a1 < 0 ? 0 : (a1 > NFc - 1 ? NFc - 1 : a1);
    sh_f1[lane] = b * NFc + f1;
    sh_f2[lane] = b * NFc + f2;
    sh_v[lane] = vm;
  }
  __syncthreads();

  const int m = lo;
  const int grow = row0 + m;
  const float* Erow  = E + (size_t)grow * Dd;
  const float* G1row = F + (size_t)sh_f1[m] * Dd;
  const float* G2row = F + (size_t)sh_f2[m] * Dd;

  // Build cat_e = [E, g1, g2] (16x384) as 12 bf16 A-fragments; stage E tile f32 to LDS
  u32x8 catf[12];
  #pragma unroll
  for (int s = 0; s < 12; ++s) {
    const float* src = (s < 4) ? Erow : (s < 8) ? G1row : G2row;
    const int kb = (s & 3) * 32;
    #pragma unroll
    for (int j = 0; j < 8; ++j) {
      int kl = a_kloc(j, hi);
      float2 p = *(const float2*)(src + kb + kl);
      catf[s][j] = pack2(p.x, p.y);
      if (s < 4) { sh_E[m * Dd + kb + kl] = p.x; sh_E[m * Dd + kb + kl + 1] = p.y; }
    }
  }
  __syncthreads();

  // GEMM1: h = GELU(cat_e @ W_fe1 + b)  (16x384)x(384x256)
  for (int nt = 0; nt < 16; ++nt) {
    f32x8 acc = {0.f,0.f,0.f,0.f,0.f,0.f,0.f,0.f};
    const unsigned short* wb = Wt_fe1 + (size_t)(nt * 16 + lo) * 384 + hi * 16;
    #pragma unroll
    for (int s = 0; s < 12; ++s)
      acc = wmma_bf16(catf[s], load_bfrag(wb + s * 32), acc);
    const float bias = b_fe1[nt * 16 + lo];
    #pragma unroll
    for (int r = 0; r < 8; ++r)
      sh_h[(hi * 8 + r) * 256 + nt * 16 + lo] = f2bf(gelu_exact(acc[r] + bias));
  }
  __syncthreads();

  // GEMM2: msg_e = (h @ W_fe2 + b) * vm  (16x256)x(256x128)
  for (int nt = 0; nt < 8; ++nt) {
    f32x8 acc = {0.f,0.f,0.f,0.f,0.f,0.f,0.f,0.f};
    const unsigned short* wb = Wt_fe2 + (size_t)(nt * 16 + lo) * 256 + hi * 16;
    #pragma unroll
    for (int s = 0; s < 8; ++s) {
      u32x8 au;
      #pragma unroll
      for (int j = 0; j < 8; ++j)
        au[j] = *(const unsigned int*)(sh_h + m * 256 + s * 32 + a_kloc(j, hi));
      acc = wmma_bf16(au, load_bfrag(wb + s * 32), acc);
    }
    const float bias = b_fe2[nt * 16 + lo];
    #pragma unroll
    for (int r = 0; r < 8; ++r) {
      int mr = hi * 8 + r;
      sh_m[mr * Dd + nt * 16 + lo] = (acc[r] + bias) * sh_v[mr];
    }
  }
  __syncthreads();

  // GEMM3: gate = sigmoid([E, msg] @ W_ge + b); x = E + gate*msg
  u32x8 gf[8];
  #pragma unroll
  for (int s = 0; s < 8; ++s) {
    const float* src = (s < 4) ? (sh_E + m * Dd + s * 32) : (sh_m + m * Dd + (s - 4) * 32);
    #pragma unroll
    for (int j = 0; j < 8; ++j) {
      int kl = a_kloc(j, hi);
      gf[s][j] = pack2(src[kl], src[kl + 1]);
    }
  }
  for (int nt = 0; nt < 8; ++nt) {
    f32x8 acc = {0.f,0.f,0.f,0.f,0.f,0.f,0.f,0.f};
    const unsigned short* wb = Wt_ge + (size_t)(nt * 16 + lo) * 256 + hi * 16;
    #pragma unroll
    for (int s = 0; s < 8; ++s)
      acc = wmma_bf16(gf[s], load_bfrag(wb + s * 32), acc);
    const float bias = b_ge[nt * 16 + lo];
    #pragma unroll
    for (int r = 0; r < 8; ++r) {
      int mr = hi * 8 + r, col = nt * 16 + lo;
      float gate = sigmoidf_(acc[r] + bias);
      sh_x[mr * Dd + col] = sh_E[mr * Dd + col] + gate * sh_m[mr * Dd + col];
    }
  }
  __syncthreads();

  // LayerNorm per row (2 lanes/row), write E_new, scatter-add to faces
  {
    const int mr = lo, half = hi;
    const float* xr = sh_x + mr * Dd;
    float s1 = 0.f, s2 = 0.f;
    for (int c = half * 64; c < half * 64 + 64; ++c) { float v = xr[c]; s1 += v; s2 += v * v; }
    s1 += __shfl_xor(s1, 16);
    s2 += __shfl_xor(s2, 16);
    float mean = s1 * (1.f / 128.f);
    float var  = s2 * (1.f / 128.f) - mean * mean;
    float rstd = rsqrtf(var + 1e-5f);
    float vm = sh_v[mr];
    float* eo  = E_out + (size_t)(row0 + mr) * Dd;
    float* fm1 = face_msg + (size_t)sh_f1[mr] * Dd;
    float* fm2 = face_msg + (size_t)sh_f2[mr] * Dd;
    for (int c = half * 64; c < half * 64 + 64; ++c) {
      float en = (xr[c] - mean) * rstd * ln_g[c] + ln_b[c];
      eo[c] = en;
      if (vm > 0.f) { atomicAdd(fm1 + c, en); atomicAdd(fm2 + c, en); }
    }
    if (half == 0 && vm > 0.f) {
      atomicAdd(face_cnt + sh_f1[mr], 1.f);
      atomicAdd(face_cnt + sh_f2[mr], 1.f);
    }
  }
}

// ---------------- face kernel: one wave per 16-face tile ----------------
__global__ __launch_bounds__(32) void eml_face_kernel(
    const float* __restrict__ F, const float* __restrict__ face_mask,
    const float* __restrict__ face_msg, const float* __restrict__ face_cnt,
    const unsigned short* __restrict__ Wt_ef1, const float* __restrict__ b_ef1,
    const unsigned short* __restrict__ Wt_ef2, const float* __restrict__ b_ef2,
    const unsigned short* __restrict__ Wt_gf,  const float* __restrict__ b_gf,
    const float* __restrict__ ln_g, const float* __restrict__ ln_b,
    float* __restrict__ F_out)
{
  __shared__ __align__(16) unsigned char smem[SMEM_BYTES];
  unsigned short* sh_h = (unsigned short*)smem;      // 16x256 bf16, reused as sh_x
  float* sh_x = (float*)smem;
  float* sh_F = (float*)(smem + 8192);               // 16x128 f32
  float* sh_m = (float*)(smem + 16384);              // 16x128 f32
  float* sh_inv  = (float*)(smem + 24576);
  float* sh_mask = (float*)(smem + 24640);

  const int tile = blockIdx.x;
  const int lane = threadIdx.x;
  const int lo = lane & 15, hi = lane >> 4;
  const int row0 = tile * 16;

  if (lane < 16) {
    sh_inv[lane]  = 1.f / (face_cnt[row0 + lane] + 1e-8f);
    sh_mask[lane] = face_mask[row0 + lane];
  }
  __syncthreads();

  const int m = lo;
  const int grow = row0 + m;                         // flattened (B*NF)
  const float* Frow = F + (size_t)grow * Dd;
  const float* Mrow = face_msg + (size_t)grow * Dd;
  const float  inv  = sh_inv[m];

  // cat_f = [F, face_msg/cnt] (16x256) as 8 A-fragments; stage F tile f32
  u32x8 catf[8];
  #pragma unroll
  for (int s = 0; s < 8; ++s) {
    const int kb = (s & 3) * 32;
    #pragma unroll
    for (int j = 0; j < 8; ++j) {
      int kl = a_kloc(j, hi);
      if (s < 4) {
        float2 p = *(const float2*)(Frow + kb + kl);
        catf[s][j] = pack2(p.x, p.y);
        sh_F[m * Dd + kb + kl] = p.x; sh_F[m * Dd + kb + kl + 1] = p.y;
      } else {
        float2 p = *(const float2*)(Mrow + kb + kl);
        catf[s][j] = pack2(p.x * inv, p.y * inv);
      }
    }
  }
  __syncthreads();

  // GEMM1: h2 = GELU(cat_f @ W_ef1 + b)  (16x256)x(256x256)
  for (int nt = 0; nt < 16; ++nt) {
    f32x8 acc = {0.f,0.f,0.f,0.f,0.f,0.f,0.f,0.f};
    const unsigned short* wb = Wt_ef1 + (size_t)(nt * 16 + lo) * 256 + hi * 16;
    #pragma unroll
    for (int s = 0; s < 8; ++s)
      acc = wmma_bf16(catf[s], load_bfrag(wb + s * 32), acc);
    const float bias = b_ef1[nt * 16 + lo];
    #pragma unroll
    for (int r = 0; r < 8; ++r)
      sh_h[(hi * 8 + r) * 256 + nt * 16 + lo] = f2bf(gelu_exact(acc[r] + bias));
  }
  __syncthreads();

  // GEMM2: msg_f = h2 @ W_ef2 + b
  for (int nt = 0; nt < 8; ++nt) {
    f32x8 acc = {0.f,0.f,0.f,0.f,0.f,0.f,0.f,0.f};
    const unsigned short* wb = Wt_ef2 + (size_t)(nt * 16 + lo) * 256 + hi * 16;
    #pragma unroll
    for (int s = 0; s < 8; ++s) {
      u32x8 au;
      #pragma unroll
      for (int j = 0; j < 8; ++j)
        au[j] = *(const unsigned int*)(sh_h + m * 256 + s * 32 + a_kloc(j, hi));
      acc = wmma_bf16(au, load_bfrag(wb + s * 32), acc);
    }
    const float bias = b_ef2[nt * 16 + lo];
    #pragma unroll
    for (int r = 0; r < 8; ++r)
      sh_m[(hi * 8 + r) * Dd + nt * 16 + lo] = acc[r] + bias;
  }
  __syncthreads();

  // GEMM3: gate_f = sigmoid([F, msg_f] @ W_gf + b); x = F + gate*msg*mask
  u32x8 gf[8];
  #pragma unroll
  for (int s = 0; s < 8; ++s) {
    const float* src = (s < 4) ? (sh_F + m * Dd + s * 32) : (sh_m + m * Dd + (s - 4) * 32);
    #pragma unroll
    for (int j = 0; j < 8; ++j) {
      int kl = a_kloc(j, hi);
      gf[s][j] = pack2(src[kl], src[kl + 1]);
    }
  }
  for (int nt = 0; nt < 8; ++nt) {
    f32x8 acc = {0.f,0.f,0.f,0.f,0.f,0.f,0.f,0.f};
    const unsigned short* wb = Wt_gf + (size_t)(nt * 16 + lo) * 256 + hi * 16;
    #pragma unroll
    for (int s = 0; s < 8; ++s)
      acc = wmma_bf16(gf[s], load_bfrag(wb + s * 32), acc);
    const float bias = b_gf[nt * 16 + lo];
    #pragma unroll
    for (int r = 0; r < 8; ++r) {
      int mr = hi * 8 + r, col = nt * 16 + lo;
      float gate = sigmoidf_(acc[r] + bias);
      sh_x[mr * Dd + col] = sh_F[mr * Dd + col] + gate * sh_m[mr * Dd + col] * sh_mask[mr];
    }
  }
  __syncthreads();

  // LayerNorm + write F_new
  {
    const int mr = lo, half = hi;
    const float* xr = sh_x + mr * Dd;
    float s1 = 0.f, s2 = 0.f;
    for (int c = half * 64; c < half * 64 + 64; ++c) { float v = xr[c]; s1 += v; s2 += v * v; }
    s1 += __shfl_xor(s1, 16);
    s2 += __shfl_xor(s2, 16);
    float mean = s1 * (1.f / 128.f);
    float var  = s2 * (1.f / 128.f) - mean * mean;
    float rstd = rsqrtf(var + 1e-5f);
    float* fo = F_out + (size_t)(row0 + mr) * Dd;
    for (int c = half * 64; c < half * 64 + 64; ++c)
      fo[c] = (xr[c] - mean) * rstd * ln_g[c] + ln_b[c];
  }
}

extern "C" void kernel_launch(void* const* d_in, const int* in_sizes, int n_in,
                              void* d_out, int out_size, void* d_ws, size_t ws_size,
                              hipStream_t stream) {
  (void)in_sizes; (void)n_in; (void)out_size; (void)ws_size;
  const float* F         = (const float*)d_in[0];
  const float* E         = (const float*)d_in[1];
  const int*   etf       = (const int*)d_in[2];
  const float* face_mask = (const float*)d_in[3];
  const float* edge_mask = (const float*)d_in[4];
  const float* W_fe1 = (const float*)d_in[5];  const float* b_fe1 = (const float*)d_in[6];
  const float* W_fe2 = (const float*)d_in[7];  const float* b_fe2 = (const float*)d_in[8];
  const float* W_ef1 = (const float*)d_in[9];  const float* b_ef1 = (const float*)d_in[10];
  const float* W_ef2 = (const float*)d_in[11]; const float* b_ef2 = (const float*)d_in[12];
  const float* W_ge  = (const float*)d_in[13]; const float* b_ge  = (const float*)d_in[14];
  const float* W_gf  = (const float*)d_in[15]; const float* b_gf  = (const float*)d_in[16];
  const float* ln_e_g = (const float*)d_in[17]; const float* ln_e_b = (const float*)d_in[18];
  const float* ln_f_g = (const float*)d_in[19]; const float* ln_f_b = (const float*)d_in[20];

  float* out_F = (float*)d_out;
  float* out_E = out_F + (size_t)Bc * NFc * Dd;

  char* ws = (char*)d_ws;
  float* fmsg = (float*)ws;                                           // B*NF*D f32
  float* fcnt = (float*)(ws + (size_t)Bc * NFc * Dd * 4);             // B*NF f32
  unsigned short* wt = (unsigned short*)(ws + (size_t)Bc * NFc * Dd * 4 + (size_t)Bc * NFc * 4);
  unsigned short* Wt_fe1 = wt;                   // [256][384]
  unsigned short* Wt_fe2 = Wt_fe1 + 384 * 256;   // [128][256]
  unsigned short* Wt_ge  = Wt_fe2 + 256 * 128;   // [128][256]
  unsigned short* Wt_ef1 = Wt_ge  + 256 * 128;   // [256][256]
  unsigned short* Wt_ef2 = Wt_ef1 + 256 * 256;   // [128][256]
  unsigned short* Wt_gf  = Wt_ef2 + 256 * 128;   // [128][256]

  hipMemsetAsync(fmsg, 0, (size_t)(Bc * NFc * Dd + Bc * NFc) * 4, stream);

  eml_wconv<<<(384 * 256 + 255) / 256, 256, 0, stream>>>(W_fe1, Wt_fe1, 384, 256);
  eml_wconv<<<(256 * 128 + 255) / 256, 256, 0, stream>>>(W_fe2, Wt_fe2, 256, 128);
  eml_wconv<<<(256 * 128 + 255) / 256, 256, 0, stream>>>(W_ge,  Wt_ge,  256, 128);
  eml_wconv<<<(256 * 256 + 255) / 256, 256, 0, stream>>>(W_ef1, Wt_ef1, 256, 256);
  eml_wconv<<<(256 * 128 + 255) / 256, 256, 0, stream>>>(W_ef2, Wt_ef2, 256, 128);
  eml_wconv<<<(256 * 128 + 255) / 256, 256, 0, stream>>>(W_gf,  Wt_gf,  256, 128);

  eml_edge_kernel<<<(Bc * NEc) / 16, 32, 0, stream>>>(
      F, E, etf, edge_mask, Wt_fe1, b_fe1, Wt_fe2, b_fe2, Wt_ge, b_ge,
      ln_e_g, ln_e_b, out_E, fmsg, fcnt);

  eml_face_kernel<<<(Bc * NFc) / 16, 32, 0, stream>>>(
      F, face_mask, fmsg, fcnt, Wt_ef1, b_ef1, Wt_ef2, b_ef2, Wt_gf, b_gf,
      ln_f_g, ln_f_b, out_F);
}
